// ImplicitCoTModelWithRNN_86552180949517
// MI455X (gfx1250) — matrix operand order repaired and checked by reading, not compiled, not run on hardware
//
#include <hip/hip_runtime.h>
#include <hip/hip_bf16.h>

#define H_ 1024
#define S_ 256
#define B_ 4

typedef __attribute__((ext_vector_type(16))) __bf16 v16bf;
typedef __attribute__((ext_vector_type(8)))  __bf16 v8bf;
typedef __attribute__((ext_vector_type(8)))  float  v8f;

__device__ __forceinline__ __bf16 to_bf16(float f) {
    unsigned u = __builtin_bit_cast(unsigned, f);
    u += 0x7FFFu + ((u >> 16) & 1u);          // round-to-nearest-even
    unsigned short h = (unsigned short)(u >> 16);
    return __builtin_bit_cast(__bf16, h);
}

// A fragment (16x32 bf16) from bf16 row-major A (lda = Ktot).
// lane L: M = L&15 ; half = L>>4 selects K = half*16 .. +15, contiguous
__device__ __forceinline__ v16bf load_a_frag(const __bf16* __restrict__ A, int lda,
                                             int m, int kbase, int half) {
    const __bf16* p = A + (size_t)m * lda + kbase;
    v8bf lo = *(const v8bf*)(p + half * 16);
    v8bf hi = *(const v8bf*)(p + half * 16 + 8);
    v16bf a;
#pragma unroll
    for (int j = 0; j < 8; ++j) { a[j] = lo[j]; a[8 + j] = hi[j]; }
    return a;
}

// B fragment (32x16 bf16) from bf16 row-major W(N,K) (ldb = Ktot).
// lane L: N = n0 + (L&15) ; K = kbase + (L>>4)*16 .. +15, contiguous 32B.
__device__ __forceinline__ v16bf load_b_frag(const __bf16* __restrict__ W, int ldb,
                                             int n, int kbase, int half) {
    return *(const v16bf*)(W + (size_t)n * ldb + kbase + half * 16);
}

// ---------------------------------------------------------------------------
// Generic small-M GEMM, all-bf16 operands:
//   C(16xN) = act( A1@W1^T [+ A2@W2^T] + b1 [+ b2] [+ addend] )
// Stores fp32 (Cf) and/or bf16 (Cbf); optional bf16 side-store of rows 0..3.
// One wave per 16-wide N tile, 8 waves per block.
// ---------------------------------------------------------------------------
__global__ void gemm_wmma(const __bf16* __restrict__ A1, const __bf16* __restrict__ W1, int K1,
                          const __bf16* __restrict__ A2, const __bf16* __restrict__ W2, int K2,
                          const float* __restrict__ bias1, const float* __restrict__ bias2,
                          const float* __restrict__ addend, int addend_ld,
                          float* __restrict__ Cf, __bf16* __restrict__ Cbf, int ldc,
                          __bf16* __restrict__ out2, long long out2_stride,
                          int N, int relu) {
    int wave = threadIdx.x >> 5;
    int lane = threadIdx.x & 31;
    int tile = blockIdx.x * (blockDim.x >> 5) + wave;
    if (tile * 16 >= N) return;                 // wave-uniform; EXEC stays all-1
    int nl = lane & 15;
    int half = lane >> 4;
    int n = tile * 16 + nl;

    v8f acc = {};
    for (int k = 0; k < K1; k += 32) {
        v16bf a = load_a_frag(A1, K1, nl, k, half);
        v16bf b = load_b_frag(W1, K1, n, k, half);
        acc = __builtin_amdgcn_wmma_f32_16x16x32_bf16(false, a, false, b,
                                                      (short)0, acc, false, false);
    }
    if (A2) {
        for (int k = 0; k < K2; k += 32) {
            v16bf a = load_a_frag(A2, K2, nl, k, half);
            v16bf b = load_b_frag(W2, K2, n, k, half);
            acc = __builtin_amdgcn_wmma_f32_16x16x32_bf16(false, a, false, b,
                                                          (short)0, acc, false, false);
        }
    }

    float bsum = 0.f;
    if (bias1) bsum += bias1[n];
    if (bias2) bsum += bias2[n];
#pragma unroll
    for (int r = 0; r < 8; ++r) {
        int m = r + half * 8;                   // C layout: VGPR r -> M = r + 8*(lane>>4)
        float v = acc[r] + bsum;
        if (addend) v += addend[(size_t)m * addend_ld + n];
        if (relu) v = fmaxf(v, 0.f);
        if (Cf)  Cf[(size_t)m * ldc + n] = v;
        if (Cbf) Cbf[(size_t)m * ldc + n] = to_bf16(v);
        if (out2 && m < B_) out2[(size_t)m * out2_stride + n] = to_bf16(v);
    }
}

// ---------------------------------------------------------------------------
// LSTM pointwise: gates (16,4H) fp32 -> c fp32, h bf16; h also into aoutbf[:, :H]
// ---------------------------------------------------------------------------
__global__ void lstm_pointwise(const float* __restrict__ g,
                               float* __restrict__ c, __bf16* __restrict__ hbf,
                               __bf16* __restrict__ aoutbf) {
    int idx = blockIdx.x * blockDim.x + threadIdx.x;
    if (idx >= 16 * H_) return;
    int m = idx / H_, k = idx - m * H_;
    const float* gr = g + (size_t)m * 4 * H_;
    float gi = gr[k], gf = gr[H_ + k], gg = gr[2 * H_ + k], go = gr[3 * H_ + k];
    float si = 1.f / (1.f + __expf(-gi));
    float sf = 1.f / (1.f + __expf(-gf));
    float so = 1.f / (1.f + __expf(-go));
    float cn = sf * c[idx] + si * tanhf(gg);
    float hn = so * tanhf(cn);
    c[idx] = cn;
    __bf16 hb = to_bf16(hn);
    hbf[idx] = hb;
    aoutbf[(size_t)m * 2 * H_ + k] = hb;
}

// ---------------------------------------------------------------------------
// Attention step i: scores over keys[j<i], softmax, ctx; append cur_key at i.
// One block per batch element. ctx stored fp32 (addend) + bf16 (aout[:, H:]).
// ---------------------------------------------------------------------------
__global__ void attention_step(float* __restrict__ keys,        // (B,S,H) fp32
                               const float* __restrict__ kq,    // (16,2H) fp32: key|query
                               float* __restrict__ ctx,         // (16,H) fp32
                               __bf16* __restrict__ aoutbf,     // (16,2H) bf16
                               int i) {
    __shared__ float q[H_];
    __shared__ float red[256];
    __shared__ float probs[256];
    int b = blockIdx.x, t = threadIdx.x;

    for (int k = t; k < H_; k += 256) q[k] = kq[(size_t)b * 2 * H_ + H_ + k];
    __syncthreads();

    float sc = -3.0e38f;
    if (t < i) {
        const float* kr = keys + ((size_t)b * S_ + t) * H_;
        float s = 0.f;
        for (int k = 0; k < H_; ++k) s += kr[k] * q[k];
        sc = s;
    }
    red[t] = sc; __syncthreads();
    for (int off = 128; off > 0; off >>= 1) {
        if (t < off) red[t] = fmaxf(red[t], red[t + off]);
        __syncthreads();
    }
    float mx = red[0]; __syncthreads();
    float e = (t < i) ? __expf(sc - mx) : 0.f;
    red[t] = e; __syncthreads();
    for (int off = 128; off > 0; off >>= 1) {
        if (t < off) red[t] += red[t + off];
        __syncthreads();
    }
    float denom = red[0];
    probs[t] = (i > 0) ? (e / denom) : 0.f;
    __syncthreads();

    for (int k = t; k < H_; k += 256) {
        float acc = 0.f;
        for (int j = 0; j < i; ++j) acc += probs[j] * keys[((size_t)b * S_ + j) * H_ + k];
        ctx[(size_t)b * H_ + k] = acc;
        aoutbf[(size_t)b * 2 * H_ + H_ + k] = to_bf16(acc);
    }
    for (int k = t; k < H_; k += 256)
        keys[((size_t)b * S_ + i) * H_ + k] = kq[(size_t)b * 2 * H_ + k];
}

// ---------------------------------------------------------------------------
// LM head: A (M=1024,K=1024) bf16 @ lm_W(N=32000,K)^T, 4 M-tiles per wave.
// Wbf preferred (pure loads); Wf fallback converts in-register.
// ---------------------------------------------------------------------------
#define LM_MT 4
__global__ void lm_head(const __bf16* __restrict__ A,
                        const __bf16* __restrict__ Wbf,
                        const float* __restrict__ Wf,
                        float* __restrict__ out, int M, int N, int K) {
    int wave = threadIdx.x >> 5;
    int lane = threadIdx.x & 31;
    int ntile = blockIdx.x * (blockDim.x >> 5) + wave;
    int mgroup = blockIdx.y;
    if (ntile * 16 >= N) return;
    int nl = lane & 15;
    int half = lane >> 4;
    int n = ntile * 16 + nl;

    v8f acc[LM_MT];
#pragma unroll
    for (int mt = 0; mt < LM_MT; ++mt) acc[mt] = (v8f){};

    if (Wbf) {
        for (int k = 0; k < K; k += 32) {
            v16bf bfr = load_b_frag(Wbf, K, n, k, half);
#pragma unroll
            for (int mt = 0; mt < LM_MT; ++mt) {
                int mbase = (mgroup * LM_MT + mt) * 16;
                v16bf a = load_a_frag(A + (size_t)mbase * K, K, nl, k, half);
                acc[mt] = __builtin_amdgcn_wmma_f32_16x16x32_bf16(
                    false, a, false, bfr, (short)0, acc[mt], false, false);
            }
        }
    } else {
        for (int k = 0; k < K; k += 32) {
            const float* wp = Wf + (size_t)n * K + k + half * 16;
            v16bf bfr;
#pragma unroll
            for (int j = 0; j < 16; ++j) bfr[j] = to_bf16(wp[j]);
#pragma unroll
            for (int mt = 0; mt < LM_MT; ++mt) {
                int mbase = (mgroup * LM_MT + mt) * 16;
                v16bf a = load_a_frag(A + (size_t)mbase * K, K, nl, k, half);
                acc[mt] = __builtin_amdgcn_wmma_f32_16x16x32_bf16(
                    false, a, false, bfr, (short)0, acc[mt], false, false);
            }
        }
    }
#pragma unroll
    for (int mt = 0; mt < LM_MT; ++mt) {
#pragma unroll
        for (int r = 0; r < 8; ++r) {
            int row = (mgroup * LM_MT + mt) * 16 + r + half * 8;
            out[(size_t)row * N + n] = acc[mt][r];
        }
    }
}

// ---------------------------------------------------------------------------
// Setup kernels
// ---------------------------------------------------------------------------
__global__ void f32_to_bf16_k(const float* __restrict__ src, __bf16* __restrict__ dst, long long n) {
    long long i = (long long)blockIdx.x * blockDim.x + threadIdx.x;
    long long stride = (long long)gridDim.x * blockDim.x;
    for (; i < n; i += stride) dst[i] = to_bf16(src[i]);
}
__global__ void zero_f32(float* p, long long n) {
    long long i = (long long)blockIdx.x * blockDim.x + threadIdx.x;
    long long stride = (long long)gridDim.x * blockDim.x;
    for (; i < n; i += stride) p[i] = 0.f;
}
__global__ void init_a0_k(__bf16* __restrict__ a0bf, const float* __restrict__ hs,
                          const float* __restrict__ mix) {
    int idx = blockIdx.x * blockDim.x + threadIdx.x;
    if (idx >= 16 * 2 * H_) return;
    int m = idx / (2 * H_), k = idx - m * 2 * H_;
    float v;
    if (k < H_) v = (m < B_) ? hs[(size_t)m * S_ * H_ + k] : 0.f;
    else        v = mix[k - H_];
    a0bf[idx] = to_bf16(v);
}
__global__ void init_h0_k(__bf16* __restrict__ hfbf, const float* __restrict__ hs) {
    int idx = blockIdx.x * blockDim.x + threadIdx.x;
    if (idx >= B_ * H_) return;
    int b = idx / H_, k = idx - b * H_;
    hfbf[(size_t)b * S_ * H_ + k] = to_bf16(hs[(size_t)b * S_ * H_ + k]);
}
__global__ void build_kqb_k(float* __restrict__ kqb, const float* __restrict__ kb,
                            const float* __restrict__ qb) {
    int idx = blockIdx.x * blockDim.x + threadIdx.x;
    if (idx >= 2 * H_) return;
    kqb[idx] = (idx < H_) ? kb[idx] : qb[idx - H_];
}

// ---------------------------------------------------------------------------
extern "C" void kernel_launch(void* const* d_in, const int* in_sizes, int n_in,
                              void* d_out, int out_size, void* d_ws, size_t ws_size,
                              hipStream_t stream) {
    const float* hs      = (const float*)d_in[0];
    const float* mixture = (const float*)d_in[1];
    const float* mlp_W1  = (const float*)d_in[2];
    const float* mlp_b1  = (const float*)d_in[3];
    const float* mlp_W2  = (const float*)d_in[4];
    const float* mlp_b2  = (const float*)d_in[5];
    const float* W_ih    = (const float*)d_in[6];
    const float* W_hh    = (const float*)d_in[7];
    const float* b_ih    = (const float*)d_in[8];
    const float* b_hh    = (const float*)d_in[9];
    const float* key_W   = (const float*)d_in[10];
    const float* key_b   = (const float*)d_in[11];
    const float* query_W = (const float*)d_in[12];
    const float* query_b = (const float*)d_in[13];
    const float* out_W   = (const float*)d_in[14];
    const float* out_b   = (const float*)d_in[15];
    const float* lm_W    = (const float*)d_in[16];

    const int H = H_, S = S_, V = 32000;
    const int H2 = 2 * H, H4 = 4 * H;

    char* ws = (char*)d_ws;
    size_t off = 0;
    auto alloc = [&](size_t bytes) -> char* {
        char* p = ws + off;
        off = (off + bytes + 255) & ~(size_t)255;
        return p;
    };

    // bf16 weight cache (converted every call; deterministic)
    __bf16* w1bf  = (__bf16*)alloc((size_t)H4 * H2 * 2);
    __bf16* w2bf  = (__bf16*)alloc((size_t)H  * H4 * 2);
    __bf16* wihbf = (__bf16*)alloc((size_t)H4 * H  * 2);
    __bf16* whhbf = (__bf16*)alloc((size_t)H4 * H  * 2);
    __bf16* kqwbf = (__bf16*)alloc((size_t)H2 * H  * 2);   // [key_W ; query_W]
    __bf16* outwbf= (__bf16*)alloc((size_t)H  * H2 * 2);

    // bf16 activation mirrors
    __bf16* a0bf  = (__bf16*)alloc((size_t)16 * H2 * 2);
    __bf16* h1bf  = (__bf16*)alloc((size_t)16 * H4 * 2);
    __bf16* fxbf  = (__bf16*)alloc((size_t)16 * H * 2);
    __bf16* hbf   = (__bf16*)alloc((size_t)16 * H * 2);
    __bf16* hfbf  = (__bf16*)alloc((size_t)B_ * S * H * 2);

    // fp32 + zero-initialized state block (includes aoutbf so rows 4..15 stay 0)
    size_t state_begin = off;
    float*  keys   = (float*)alloc((size_t)B_ * S * H * 4);
    float*  ctx    = (float*)alloc((size_t)16 * H * 4);
    float*  cbuf   = (float*)alloc((size_t)16 * H * 4);
    float*  g      = (float*)alloc((size_t)16 * H4 * 4);
    float*  kqbuf  = (float*)alloc((size_t)16 * H2 * 4);
    float*  kqb    = (float*)alloc((size_t)H2 * 4);
    __bf16* aoutbf = (__bf16*)alloc((size_t)16 * H2 * 2);
    size_t state_end = off;

    // bf16 lm_W copy last; fall back if workspace too small
    __bf16* wlmbf = (__bf16*)alloc((size_t)V * H * 2);
    bool use_lm_bf16 = (off <= ws_size);
    (void)in_sizes; (void)n_in; (void)out_size;

    // ---- setup ----
    f32_to_bf16_k<<<1024, 256, 0, stream>>>(mlp_W1, w1bf,  (long long)H4 * H2);
    f32_to_bf16_k<<<1024, 256, 0, stream>>>(mlp_W2, w2bf,  (long long)H  * H4);
    f32_to_bf16_k<<<1024, 256, 0, stream>>>(W_ih,   wihbf, (long long)H4 * H);
    f32_to_bf16_k<<<1024, 256, 0, stream>>>(W_hh,   whhbf, (long long)H4 * H);
    f32_to_bf16_k<<<512,  256, 0, stream>>>(key_W,   kqwbf,                 (long long)H * H);
    f32_to_bf16_k<<<512,  256, 0, stream>>>(query_W, kqwbf + (size_t)H * H, (long long)H * H);
    f32_to_bf16_k<<<1024, 256, 0, stream>>>(out_W,  outwbf,(long long)H * H2);
    if (use_lm_bf16)
        f32_to_bf16_k<<<4096, 256, 0, stream>>>(lm_W, wlmbf, (long long)V * H);

    zero_f32<<<2048, 256, 0, stream>>>((float*)(ws + state_begin),
                                       (long long)((state_end - state_begin) / 4));
    init_a0_k<<<(16 * H2 + 255) / 256, 256, 0, stream>>>(a0bf, hs, mixture);
    init_h0_k<<<(B_ * H + 255) / 256, 256, 0, stream>>>(hfbf, hs);
    build_kqb_k<<<(H2 + 255) / 256, 256, 0, stream>>>(kqb, key_b, query_b);

    auto gblocks = [](int N) { return dim3((unsigned)((N / 16 + 7) / 8)); };

    // ---- recurrence: steps 0..S-2 (step S-1's output is discarded by the ref) ----
    for (int i = 0; i < S - 1; ++i) {
        // h1 = relu(a0 @ mlp_W1^T + b1)   -> bf16 only
        gemm_wmma<<<gblocks(H4), 256, 0, stream>>>(
            a0bf, w1bf, H2, nullptr, nullptr, 0, mlp_b1, nullptr,
            nullptr, 0, nullptr, h1bf, H4, nullptr, 0, H4, 1);
        // fx = h1 @ mlp_W2^T + b2 + ctx   -> bf16 only
        gemm_wmma<<<gblocks(H), 256, 0, stream>>>(
            h1bf, w2bf, H4, nullptr, nullptr, 0, mlp_b2, nullptr,
            ctx, H, nullptr, fxbf, H, nullptr, 0, H, 0);
        // g = fx @ W_ih^T + h @ W_hh^T + b_ih + b_hh  -> fp32 (gates)
        gemm_wmma<<<gblocks(H4), 256, 0, stream>>>(
            fxbf, wihbf, H, hbf, whhbf, H, b_ih, b_hh,
            nullptr, 0, g, nullptr, H4, nullptr, 0, H4, 0);
        // gates -> c fp32, h bf16 ; h into aout[:, :H]
        lstm_pointwise<<<(16 * H + 255) / 256, 256, 0, stream>>>(g, cbuf, hbf, aoutbf);
        // kq = h @ [key_W;query_W]^T + [key_b;query_b]  -> fp32 (attention)
        gemm_wmma<<<gblocks(H2), 256, 0, stream>>>(
            hbf, kqwbf, H, nullptr, nullptr, 0, kqb, nullptr,
            nullptr, 0, kqbuf, nullptr, H2, nullptr, 0, H2, 0);
        // attention: ctx fp32 + bf16 into aout[:, H:] ; append key[i]
        attention_step<<<B_, 256, 0, stream>>>(keys, kqbuf, ctx, aoutbf, i);
        // out = a_out @ out_W^T + out_b -> next z (a0bf[:, :H]);
        // rows 0..3 also stored to hidden_final bf16 slot i+1
        gemm_wmma<<<gblocks(H), 256, 0, stream>>>(
            aoutbf, outwbf, H2, nullptr, nullptr, 0, out_b, nullptr,
            nullptr, 0, nullptr, a0bf, H2, hfbf + (size_t)(i + 1) * H,
            (long long)S * H, H, 0);
    }

    // ---- LM head: logits = hidden_final @ lm_W^T ----
    {
        int M = B_ * S, N = V, K = H;
        dim3 grid((unsigned)((N / 16 + 7) / 8), (unsigned)(M / 16 / LM_MT));
        lm_head<<<grid, 256, 0, stream>>>(hfbf, use_lm_bf16 ? wlmbf : nullptr,
                                          lm_W, (float*)d_out, M, N, K);
    }
}
